// Eq3NetMini_30090540875888
// MI455X (gfx1250) — matrix operand order repaired and checked by reading, not compiled, not run on hardware
//
#include <hip/hip_runtime.h>
#include <hip/hip_bf16.h>

typedef __attribute__((ext_vector_type(16))) _Float16 v16h;
typedef __attribute__((ext_vector_type(8)))  _Float16 v8h;
typedef __attribute__((ext_vector_type(8)))  float    v8f;

#define BSZ   8
#define NN    48
#define ED    32
#define IND   33          // embed + xfeat
#define HID   32
#define KPAD  64          // 33 features + 1 bias-row + zero pad -> 2 K-steps of 32
#define KROW  72          // padded row length (144 B: 16B-aligned, conflict-free stride)
#define NPAIRS (NN*NN)    // 2304
#define BLK_Y 24
#define PAIRS_PER_BLOCK (NPAIRS/BLK_Y)       // 96
#define WAVES 8
#define PAIRS_PER_WAVE (PAIRS_PER_BLOCK/WAVES) // 12

__global__ void eq3_zero_acc(float* __restrict__ acc) {
    if (threadIdx.x < BSZ*HID) acc[threadIdx.x] = 0.0f;
}

__global__ __launch_bounds__(256) void eq3_main(
    const int*   __restrict__ xcat,
    const float* __restrict__ xfeat,
    const float* __restrict__ embed,
    const float* __restrict__ coefs,   // (33, 8, 32)
    const float* __restrict__ eq_bias, // (32)
    float*       __restrict__ acc)     // (B, 32)
{
    __shared__ float xs[NN][IND+1];                      // x[b] staged, padded
    __shared__ float mbar[IND+1];                        // column means over i
    __shared__ __align__(16) _Float16 xaug[NN][KROW];    // [i][k]: k<33=x, k==33=1, else 0
    __shared__ __align__(16) _Float16 Gb[WAVES][HID][KROW]; // per-wave B-matrix, TRANSPOSED [h][K]

    const int b    = blockIdx.x;
    const int tid  = threadIdx.x;
    const int lane = tid & 31;
    const int w    = tid >> 5;

    // pull the hot coef table toward WGP$/L2 (global_prefetch_b8)
    if (tid*32 < IND*8*HID) __builtin_prefetch(coefs + tid*32, 0, 3);

    // ---- stage x[b] = [embed[xcat], xfeat] into LDS ----
    for (int e = tid; e < NN*IND; e += 256) {
        int i = e / IND, d = e - i*IND;
        float v = (d < ED) ? embed[xcat[b*NN + i]*ED + d] : xfeat[b*NN + i];
        xs[i][d] = v;
    }
    __syncthreads();

    // ---- column means, f16 augmented A-matrix (incl. zero pad), zero G K-tails ----
    if (tid < IND) {
        float s = 0.0f;
        for (int i = 0; i < NN; ++i) s += xs[i][tid];
        mbar[tid] = s * (1.0f/(float)NN);
    }
    for (int e = tid; e < NN*KROW; e += 256) {
        int i = e / KROW, kk = e - i*KROW;
        float v = (kk < IND) ? xs[i][kk] : (kk == IND ? 1.0f : 0.0f);
        xaug[i][kk] = (_Float16)v;
    }
    for (int e = tid; e < WAVES*HID*(KROW-34); e += 256) {  // K cols 34..71 := 0 (once)
        int ww = e / (HID*(KROW-34));
        int r  = e % (HID*(KROW-34));
        Gb[ww][r/(KROW-34)][34 + r%(KROW-34)] = (_Float16)0.0f;
    }
    __syncthreads();

    // ---- pack per-wave A fragments once: two contiguous 16B LDS chunks per frag ----
    const int mlo  = lane & 15;
    const int klo8 = (lane < 16) ? 0 : 8;     // ISA 16-bit 16x32 A layout, lane-half K base
    v16h afr[3][2];
    for (int it = 0; it < 3; ++it)
        for (int ks = 0; ks < 2; ++ks) {
            const v8h* plo = (const v8h*)&xaug[it*16 + mlo][ks*32 + klo8];        // K klo8+0..7
            const v8h* phi = (const v8h*)&xaug[it*16 + mlo][ks*32 + 16 + klo8];   // K 16+klo8+0..7
            v8h lo = *plo, hi = *phi;
            afr[it][ks] = __builtin_shufflevector(lo, hi,
                0,1,2,3,4,5,6,7, 8,9,10,11,12,13,14,15);
        }

    // ---- per-channel constants: eq_bias and mbar^3 term (slice 7) ----
    const int h = lane;
    float cbase = eq_bias[h];
    for (int d = 0; d < IND; ++d) {
        float mb = mbar[d];
        cbase += coefs[(d*8 + 7)*HID + h] * mb*mb*mb;
    }

    float racc0 = 0.0f, racc1 = 0.0f;
    const int nlo  = lane & 15;
    const int kb16 = (lane < 16) ? 0 : 8;     // ISA 16-bit 32x16 B layout, lane-half K base

    // ---- main loop over (j,k) pairs: build G, GEMM, fused ReLU+pool ----
    for (int t = 0; t < PAIRS_PER_WAVE; ++t) {
        int p = blockIdx.y * PAIRS_PER_BLOCK + t*WAVES + w;
        int j = p / NN, k = p - j*NN;

        // G row h (lane-owned, contiguous in K): coalesced coef reads
        float Ch = cbase;
        for (int d = 0; d < IND; ++d) {
            float xj = xs[j][d], xk = xs[k][d], mb = mbar[d];
            float mb2 = mb*mb;
            float p0  = xj*xk;
            const float* cf = coefs + d*8*HID + h;
            float g = cf[0*HID]*p0 + cf[2*HID]*(mb*xk)
                    + cf[3*HID]*(mb*xj) + cf[6*HID]*mb2;
            Gb[w][h][d] = (_Float16)g;
            Ch += cf[1*HID]*(mb*p0) + cf[4*HID]*(mb2*xk) + cf[5*HID]*(mb2*xj);
        }
        Gb[w][h][33] = (_Float16)Ch;   // bias row, pairs with Xaug ones-column
        asm volatile("s_wait_dscnt 0" ::: "memory");   // wave-local LDS RAW fence

        v8f z = {};
        v8f c[3][2];
        for (int it = 0; it < 3; ++it) { c[it][0] = z; c[it][1] = z; }

        for (int ks = 0; ks < 2; ++ks) {
            v16h bf[2];
            for (int ht = 0; ht < 2; ++ht) {
                // lane covers col n, K kb16+0..7 and 16+kb16+0..7: 2x ds_load_b128
                const v8h* pl = (const v8h*)&Gb[w][ht*16 + nlo][ks*32 + kb16];
                const v8h* ph = (const v8h*)&Gb[w][ht*16 + nlo][ks*32 + 16 + kb16];
                v8h lo = *pl, hi = *ph;
                bf[ht] = __builtin_shufflevector(lo, hi,
                    0,1,2,3,4,5,6,7, 8,9,10,11,12,13,14,15);
            }
            for (int it = 0; it < 3; ++it)
                for (int ht = 0; ht < 2; ++ht)
                    c[it][ht] = __builtin_amdgcn_wmma_f32_16x16x32_f16(
                        false, afr[it][ks], false, bf[ht],
                        (short)0, c[it][ht], false, false);
        }

        // fused ReLU + sum over i (rows); halves combined via xor-16 shuffle
        for (int ht = 0; ht < 2; ++ht) {
            float s = 0.0f;
            for (int it = 0; it < 3; ++it) {
                v8f cc = c[it][ht];
                #pragma unroll
                for (int r = 0; r < 8; ++r) s += fmaxf(cc[r], 0.0f);
            }
            s += __shfl_xor(s, 16, 32);
            if (ht == 0) racc0 += s; else racc1 += s;
        }
    }

    if (lane < 16) {
        atomicAdd(&acc[b*HID +      lane], racc0);
        atomicAdd(&acc[b*HID + 16 + lane], racc1);
    }
}

__global__ void eq3_final(const float* __restrict__ acc,
                          const float* __restrict__ out_w,
                          const float* __restrict__ out_b,
                          float*       __restrict__ out)
{
    int tid = threadIdx.x;            // 256 = 8 waves, one wave per batch row
    int b = tid >> 5, hh = tid & 31;
    const float inv = 1.0f / (float)(NN*NN*NN);
    float v = fmaxf(acc[b*HID + hh] * inv, 0.0f) * out_w[hh];
    for (int off = 16; off > 0; off >>= 1) v += __shfl_down(v, off, 32);
    if (hh == 0) out[b] = v + out_b[0];
}

extern "C" void kernel_launch(void* const* d_in, const int* in_sizes, int n_in,
                              void* d_out, int out_size, void* d_ws, size_t ws_size,
                              hipStream_t stream) {
    const int*   xcat  = (const int*)  d_in[0];
    const float* xfeat = (const float*)d_in[1];
    const float* embed = (const float*)d_in[2];
    const float* coefs = (const float*)d_in[3];
    const float* eqb   = (const float*)d_in[4];
    const float* outw  = (const float*)d_in[5];
    const float* outb  = (const float*)d_in[6];
    float* out = (float*)d_out;
    float* acc = (float*)d_ws;        // B*HID floats of scratch

    hipLaunchKernelGGL(eq3_zero_acc, dim3(1), dim3(256), 0, stream, acc);
    hipLaunchKernelGGL(eq3_main, dim3(BSZ, BLK_Y), dim3(256), 0, stream,
                       xcat, xfeat, embed, coefs, eqb, acc);
    hipLaunchKernelGGL(eq3_final, dim3(1), dim3(256), 0, stream,
                       acc, outw, outb, out);
}
